// NonPolarizable_13872744366304
// MI455X (gfx1250) — matrix operand order repaired and checked by reading, not compiled, not run on hardware
//
#include <hip/hip_runtime.h>
#include <hip/hip_bf16.h>

typedef _Float16 h16;
typedef __attribute__((ext_vector_type(8)))  _Float16 v8h;
typedef __attribute__((ext_vector_type(16))) _Float16 v16h;
typedef __attribute__((ext_vector_type(8)))  float    v8f;

#define NPER 256

// hardware reciprocal (v_rcp_f32, ~1ulp) — avoids IEEE divide expansion
__device__ __forceinline__ float frcp(float x) { return __builtin_amdgcn_rcpf(x); }

// fast silu: feeds f16 WMMA inputs, so v_rcp_f32 accuracy is plenty
__device__ __forceinline__ float silu_f(float x) {
  return x * frcp(1.0f + __expf(-x));
}

// ---- WMMA fragment loaders (wave32; this code's contiguous K order, same for A and B) ----
// A (16x32 f16): lane m (rows 0-15) / lane m+16 hold contiguous K runs of 16:
//   lanes 0-15 : K = 0..15;  lanes 16-31 : K = 16..31
__device__ __forceinline__ v16h load_afrag(const h16* buf, int ldk, int k0, int lane) {
  int m = lane & 15, h = lane >> 4;
  v8h a = *(const v8h*)(buf + m * ldk + k0 + h * 16);
  v8h b = *(const v8h*)(buf + m * ldk + k0 + h * 16 + 8);
  return __builtin_shufflevector(a, b, 0,1,2,3,4,5,6,7,8,9,10,11,12,13,14,15);
}
// B (32x16 f16): lanes 0-15 hold K=k0..k0+15 of column n, lanes 16-31 hold K=k0+16..k0+31,
// consecutive K packed 2-per-VGPR. Weights stored [n][k] so each lane reads 16 contiguous halves.
__device__ __forceinline__ v16h load_bfrag(const h16* __restrict__ W, int wk, int n0, int k0, int lane) {
  int n = lane & 15, h = lane >> 4;
  const h16* p = W + (size_t)(n0 + n) * wk + k0 + h * 16;
  v8h a = *(const v8h*)p;
  v8h b = *(const v8h*)(p + 8);
  return __builtin_shufflevector(a, b, 0,1,2,3,4,5,6,7,8,9,10,11,12,13,14,15);
}

// ---- init / small kernels ----
__global__ void k_zero_out(float* __restrict__ out, int N, int G) {
  int i = blockIdx.x * blockDim.x + threadIdx.x;
  if (i < G)      out[i] = 0.f;                  // total_energy
  if (i < 3 * G) {
    out[G + N + i]         = 0.f;                // contributions
    out[2 * N + 4 * G + i] = 0.f;                // dipole
  }
}

__global__ void k_init_node(const float* __restrict__ attrs, const float* __restrict__ E0,
                            float* __restrict__ out, int N, int G) {
  int n = blockIdx.x * blockDim.x + threadIdx.x;
  if (n >= N) return;
  float e = 0.f;
  #pragma unroll
  for (int el = 0; el < 10; ++el) e += attrs[n * 10 + el] * E0[el];
  out[G + n] = e;             // node_energy accumulator
  out[N + 4 * G + n] = 0.f;   // q
  atomicAdd(&out[G + N + (n >> 8) * 3 + 0], e);
}

__global__ void k_init_feats(const float* __restrict__ attrs, const float* __restrict__ Wemb,
                             float* __restrict__ feats, int N) {
  int idx = blockIdx.x * blockDim.x + threadIdx.x;
  if (idx >= N * 576) return;
  int n = idx / 576, r = idx % 576, mm = r >> 6, c = r & 63;
  float v = 0.f;
  if (mm == 0) {
    #pragma unroll
    for (int el = 0; el < 10; ++el) v += attrs[n * 10 + el] * Wemb[el * 64 + c];
  }
  feats[idx] = v;
}

__global__ void k_edge(const float* __restrict__ pos, const int* __restrict__ ei,
                       float* __restrict__ sh, h16* __restrict__ ef, int E) {
  int e = blockIdx.x * blockDim.x + threadIdx.x;
  if (e >= E) return;
  int s = ei[e], d = ei[E + e];
  float dx = pos[d * 3 + 0] - pos[s * 3 + 0];
  float dy = pos[d * 3 + 1] - pos[s * 3 + 1];
  float dz = pos[d * 3 + 2] - pos[s * 3 + 2];
  float r  = sqrtf(dx * dx + dy * dy + dz * dz);
  float rc = fmaxf(r, 1e-9f);
  float inv = frcp(rc);
  float x = dx * inv, y = dy * inv, z = dz * inv;
  const float s3 = 1.7320508075688772f, s15 = 3.872983346207417f, s5 = 2.23606797749979f;
  float* o = sh + (size_t)e * 9;
  o[0] = 1.f;         o[1] = s3 * x;      o[2] = s3 * y;  o[3] = s3 * z;
  o[4] = s15 * x * y; o[5] = s15 * y * z; o[6] = 0.5f * s5 * (3.f * z * z - 1.f);
  o[7] = s15 * x * z; o[8] = 0.5f * s15 * (x * x - y * y);
  float xr = r * 0.2f;
  float x5 = xr * xr; x5 = x5 * x5 * xr;
  float env = (xr < 1.f) ? 1.f - 21.f * x5 + 35.f * x5 * xr - 15.f * x5 * xr * xr : 0.f;
  float pref = 0.6324555320336759f * env * inv;  // sqrt(2/5)
  #pragma unroll
  for (int k = 0; k < 8; ++k)
    ef[(size_t)e * 8 + k] = (h16)(pref * sinf((float)(k + 1) * 3.14159265358979f * xr));
}

// convert weights to f16, transposed to [n][k] for B-fragment loads
__global__ void k_wconv(const float* __restrict__ rW1, const float* __restrict__ rW2,
                        const float* __restrict__ rW3, const float* __restrict__ rW4,
                        const float* __restrict__ Wmix,
                        h16* __restrict__ wt1, h16* __restrict__ wt2,
                        h16* __restrict__ wt3, h16* __restrict__ wt4, h16* __restrict__ wmixh) {
  int i = blockIdx.x * blockDim.x + threadIdx.x;
  if (i < 4096) {  // wt1: [t][64][32], K padded 8->32 with zeros
    int t = i / 2048, r = i % 2048, n = r / 32, k = r % 32;
    wt1[i] = (k < 8) ? (h16)rW1[t * 512 + k * 64 + n] : (h16)0.f;
    return;
  }
  i -= 4096;
  if (i < 8192) { int t = i / 4096, r = i % 4096, n = r / 64, k = r % 64;
                  wt2[i] = (h16)rW2[t * 4096 + k * 64 + n]; return; }
  i -= 8192;
  if (i < 8192) { int t = i / 4096, r = i % 4096, n = r / 64, k = r % 64;
                  wt3[i] = (h16)rW3[t * 4096 + k * 64 + n]; return; }
  i -= 8192;
  if (i < 8192) { int t = i / 4096, r = i % 4096, n = r / 64, k = r % 64;
                  wt4[i] = (h16)rW4[t * 4096 + k * 64 + n]; return; }
  i -= 8192;
  if (i < 24576) wmixh[i] = (h16)Wmix[i];  // [t][l][d][c] native: rows d, contiguous c (=k)
}

__global__ void k_zero_agg(float* __restrict__ agg, size_t n) {
  size_t i = (size_t)blockIdx.x * blockDim.x + threadIdx.x;
  size_t stride = (size_t)gridDim.x * blockDim.x;
  for (; i < n; i += stride) agg[i] = 0.f;
}

__global__ void k_nodeW(const float* __restrict__ attrs, const float* __restrict__ w2,
                        const float* __restrict__ w3, const float* __restrict__ Wsc,
                        int t, float* __restrict__ g2, float* __restrict__ g3,
                        float* __restrict__ scv, int N) {
  int idx = blockIdx.x * blockDim.x + threadIdx.x;
  if (idx >= N * 64) return;
  int n = idx >> 6, c = idx & 63;
  const float* a = attrs + (size_t)n * 10;
  float a2 = 0.f, a3 = 0.f, as = 0.f;
  #pragma unroll
  for (int el = 0; el < 10; ++el) {
    float av = a[el];
    a2 += av * w2 [t * 640 + el * 64 + c];
    a3 += av * w3 [t * 640 + el * 64 + c];
    as += av * Wsc[t * 640 + el * 64 + c];
  }
  g2[idx] = a2; g3[idx] = a3; scv[idx] = as;
}

// ---- fused 4-layer radial MLP (WMMA) + message scatter ----
__global__ __launch_bounds__(128) void k_mlp_msg(
    const h16* __restrict__ ef16, const float* __restrict__ sh,
    const int* __restrict__ eidx, int E,
    const float* __restrict__ feats, float* __restrict__ agg,
    const h16* __restrict__ Wt1, const h16* __restrict__ Wt2,
    const h16* __restrict__ Wt3, const h16* __restrict__ Wt4) {
  __shared__ alignas(16) h16 act[4][16 * 64];
  __shared__ float shT[4][16 * 9];
  __shared__ int srcT[4][16], dstT[4][16];
  int wv = threadIdx.x >> 5, lane = threadIdx.x & 31;
  int tile = blockIdx.x * 4 + wv;
  int e0 = tile * 16;
  int m = lane & 15, h = lane >> 4;

  if (lane < 16) {
    srcT[wv][lane] = eidx[e0 + lane];
    dstT[wv][lane] = eidx[E + e0 + lane];
  }
  for (int i = lane; i < 144; i += 32) shT[wv][i] = sh[(size_t)e0 * 9 + i];
  // stage ef tile (K padded 8->32)
  #pragma unroll
  for (int j = 0; j < 16; ++j) {
    int col = h * 16 + j;
    h16 v = (col < 8) ? ef16[(size_t)(e0 + m) * 8 + col] : (h16)0.f;
    act[wv][m * 64 + col] = v;
  }
  __syncthreads();

  // layer 1: (16x32)@(32x64), K padded
  v8f acc[4];
  {
    v16h A = load_afrag(&act[wv][0], 64, 0, lane);
    #pragma unroll
    for (int nt = 0; nt < 4; ++nt) {
      v16h B = load_bfrag(Wt1, 32, nt * 16, 0, lane);
      v8f cz = {};
      acc[nt] = __builtin_amdgcn_wmma_f32_16x16x32_f16(false, A, false, B, (short)0, cz, false, false);
    }
  }
  __syncthreads();
  #pragma unroll
  for (int nt = 0; nt < 4; ++nt)
    #pragma unroll
    for (int r = 0; r < 8; ++r)
      act[wv][(r + 8 * h) * 64 + nt * 16 + m] = (h16)silu_f(acc[nt][r]);
  __syncthreads();

  // layers 2,3: (16x64)@(64x64) with SiLU
  const h16* WL[2] = {Wt2, Wt3};
  #pragma unroll
  for (int L = 0; L < 2; ++L) {
    v16h A0 = load_afrag(&act[wv][0], 64, 0, lane);
    v16h A1 = load_afrag(&act[wv][0], 64, 32, lane);
    #pragma unroll
    for (int nt = 0; nt < 4; ++nt) {
      v16h B0 = load_bfrag(WL[L], 64, nt * 16, 0, lane);
      v8f cz = {};
      cz = __builtin_amdgcn_wmma_f32_16x16x32_f16(false, A0, false, B0, (short)0, cz, false, false);
      v16h B1 = load_bfrag(WL[L], 64, nt * 16, 32, lane);
      acc[nt] = __builtin_amdgcn_wmma_f32_16x16x32_f16(false, A1, false, B1, (short)0, cz, false, false);
    }
    __syncthreads();
    #pragma unroll
    for (int nt = 0; nt < 4; ++nt)
      #pragma unroll
      for (int r = 0; r < 8; ++r)
        act[wv][(r + 8 * h) * 64 + nt * 16 + m] = (h16)silu_f(acc[nt][r]);
    __syncthreads();
  }

  // layer 4 (linear) -> R tile kept in accumulators
  {
    v16h A0 = load_afrag(&act[wv][0], 64, 0, lane);
    v16h A1 = load_afrag(&act[wv][0], 64, 32, lane);
    #pragma unroll
    for (int nt = 0; nt < 4; ++nt) {
      v16h B0 = load_bfrag(Wt4, 64, nt * 16, 0, lane);
      v8f cz = {};
      cz = __builtin_amdgcn_wmma_f32_16x16x32_f16(false, A0, false, B0, (short)0, cz, false, false);
      v16h B1 = load_bfrag(Wt4, 64, nt * 16, 32, lane);
      acc[nt] = __builtin_amdgcn_wmma_f32_16x16x32_f16(false, A1, false, B1, (short)0, cz, false, false);
    }
  }

  // message: msg[e,c,m] = feats0[src[e],c] * R[e,c] * sh[e,m] / 24 -> scatter to agg[dst]
  #pragma unroll
  for (int r = 0; r < 8; ++r) {
    int em = r + 8 * h;                 // C/D layout: M = r + 8*(lane>=16)
    int sn = srcT[wv][em], dn = dstT[wv][em];
    float s9[9];
    #pragma unroll
    for (int q2 = 0; q2 < 9; ++q2) s9[q2] = shT[wv][em * 9 + q2] * (1.0f / 24.0f);
    #pragma unroll
    for (int nt = 0; nt < 4; ++nt) {
      int c = nt * 16 + m;              // N = lane&15
      float sv = feats[(size_t)sn * 576 + c];
      float base = sv * acc[nt][r];
      #pragma unroll
      for (int q2 = 0; q2 < 9; ++q2)
        atomicAdd(&agg[(size_t)dn * 576 + q2 * 64 + c], base * s9[q2]);
    }
  }
}

// ---- per-m channel mixing (WMMA), in-place agg -> mixed ----
__global__ __launch_bounds__(128) void k_mix(float* __restrict__ agg,
                                             const h16* __restrict__ WmixH) {
  __shared__ alignas(16) h16 buf[4][16 * 64];
  int wv = threadIdx.x >> 5, lane = threadIdx.x & 31;
  int n0 = (blockIdx.x * 4 + wv) * 16;
  int m = lane & 15, h = lane >> 4;
  const int LIDX[9] = {0, 1, 1, 1, 2, 2, 2, 2, 2};
  for (int mm = 0; mm < 9; ++mm) {
    for (int i = lane; i < 1024; i += 32) {
      int row = i >> 6, col = i & 63;
      buf[wv][i] = (h16)agg[(size_t)(n0 + row) * 576 + mm * 64 + col];
    }
    __syncthreads();
    v16h A0 = load_afrag(&buf[wv][0], 64, 0, lane);
    v16h A1 = load_afrag(&buf[wv][0], 64, 32, lane);
    const h16* W = WmixH + (size_t)LIDX[mm] * 4096;  // [d][c] == [n][k]
    v8f acc[4];
    #pragma unroll
    for (int nt = 0; nt < 4; ++nt) {
      v16h B0 = load_bfrag(W, 64, nt * 16, 0, lane);
      v8f cz = {};
      cz = __builtin_amdgcn_wmma_f32_16x16x32_f16(false, A0, false, B0, (short)0, cz, false, false);
      v16h B1 = load_bfrag(W, 64, nt * 16, 32, lane);
      acc[nt] = __builtin_amdgcn_wmma_f32_16x16x32_f16(false, A1, false, B1, (short)0, cz, false, false);
    }
    __syncthreads();
    #pragma unroll
    for (int nt = 0; nt < 4; ++nt)
      #pragma unroll
      for (int r = 0; r < 8; ++r)
        agg[(size_t)(n0 + r + 8 * h) * 576 + mm * 64 + nt * 16 + m] = acc[nt][r];
    __syncthreads();
  }
}

__global__ void k_prod(const float* __restrict__ mixed, const float* __restrict__ g2,
                       const float* __restrict__ g3, const float* __restrict__ scv,
                       float* __restrict__ feats, int N) {
  int idx = blockIdx.x * blockDim.x + threadIdx.x;
  if (idx >= N * 64) return;
  int n = idx >> 6, c = idx & 63;
  size_t base = (size_t)n * 576 + c;
  float s = mixed[base];
  float fac = 1.f + g2[idx] * s + g3[idx] * s * s;
  float sc = scv[idx];
  #pragma unroll
  for (int mm = 0; mm < 9; ++mm) {
    size_t j = base + mm * 64;
    feats[j] = mixed[j] * fac + feats[j] * sc;
  }
}

__global__ void k_readout(const float* __restrict__ feats, const float* __restrict__ Wr0,
                          const float* __restrict__ Wh, const float* __restrict__ Wo,
                          const float* __restrict__ Wq, int t, float* __restrict__ out,
                          int N, int G) {
  int n = blockIdx.x * blockDim.x + threadIdx.x;
  if (n >= N) return;
  const float* f0 = feats + (size_t)n * 576;
  float ne = 0.f;
  if (t == 0) {
    for (int c = 0; c < 64; ++c) ne += f0[c] * Wr0[c];
  } else {
    #pragma unroll
    for (int j = 0; j < 16; ++j) {
      float hsum = 0.f;
      for (int c = 0; c < 64; ++c) hsum += f0[c] * Wh[c * 16 + j];
      ne += silu_f(hsum) * Wo[j];
    }
  }
  float qv = 0.f;
  for (int c = 0; c < 64; ++c) qv += f0[c] * Wq[t * 64 + c];
  out[G + n] += ne;
  out[N + 4 * G + n] += qv;
  atomicAdd(&out[G + N + (n >> 8) * 3 + (t + 1)], ne);
}

__global__ __launch_bounds__(256) void k_coulomb(const float* __restrict__ pos,
                                                 float* __restrict__ out, int N, int G) {
  __shared__ float qs[NPER], px[NPER], py[NPER], pz[NPER], red[NPER];
  int g = blockIdx.x, i = threadIdx.x;
  int n = g * NPER + i;
  const float* q = out + N + 4 * G;
  float qi = q[n];
  float x = pos[n * 3 + 0], y = pos[n * 3 + 1], z = pos[n * 3 + 2];
  qs[i] = qi; px[i] = x; py[i] = y; pz[i] = z;
  __syncthreads();
  float acc = 0.f;
  for (int j = 0; j < NPER; ++j) {
    if (j == i) continue;
    float dx = x - px[j], dy = y - py[j], dz = z - pz[j];
    float r = sqrtf(dx * dx + dy * dy + dz * dz + 1e-12f);
    acc += qs[j] * erff(0.5f * r) * frcp(r);
  }
  float vals[4] = {0.5f * qi * acc, qi * x, qi * y, qi * z};
  float res[4];
  for (int v = 0; v < 4; ++v) {
    __syncthreads();
    red[i] = vals[v];
    __syncthreads();
    for (int s = 128; s > 0; s >>= 1) {
      if (i < s) red[i] += red[i + s];
      __syncthreads();
    }
    res[v] = red[0];
  }
  if (i == 0) {
    const float* cb = out + G + N;
    out[g] = cb[g * 3 + 0] + cb[g * 3 + 1] + cb[g * 3 + 2] + res[0];
    float* dip = out + 2 * N + 4 * G;
    dip[g * 3 + 0] = res[1]; dip[g * 3 + 1] = res[2]; dip[g * 3 + 2] = res[3];
  }
}

extern "C" void kernel_launch(void* const* d_in, const int* in_sizes, int n_in,
                              void* d_out, int out_size, void* d_ws, size_t ws_size,
                              hipStream_t stream) {
  const float* attrs = (const float*)d_in[0];
  const float* pos   = (const float*)d_in[1];
  const int*   eidx  = (const int*)d_in[2];
  const float* Wemb  = (const float*)d_in[4];
  const float* E0    = (const float*)d_in[5];
  const float* rW1   = (const float*)d_in[6];
  const float* rW2   = (const float*)d_in[7];
  const float* rW3   = (const float*)d_in[8];
  const float* rW4   = (const float*)d_in[9];
  const float* Wmix  = (const float*)d_in[10];
  const float* Wsc   = (const float*)d_in[11];
  const float* w2    = (const float*)d_in[12];
  const float* w3    = (const float*)d_in[13];
  const float* Wr0   = (const float*)d_in[14];
  const float* Wh    = (const float*)d_in[15];
  const float* Wo    = (const float*)d_in[16];
  const float* Wq    = (const float*)d_in[17];
  float* out = (float*)d_out;

  int N = in_sizes[0] / 10;     // 16384
  int E = in_sizes[2] / 2;      // 393216
  int G = N / NPER;             // 64

  uint8_t* ws = (uint8_t*)d_ws;
  size_t off = 0;
  auto alloc = [&](size_t bytes) {
    void* p = ws + off;
    off += (bytes + 255) & ~(size_t)255;
    return p;
  };
  float* sh    = (float*)alloc((size_t)E * 9 * 4);
  h16*   ef16  = (h16*)  alloc((size_t)E * 8 * 2);
  float* feats = (float*)alloc((size_t)N * 576 * 4);
  float* agg   = (float*)alloc((size_t)N * 576 * 4);
  float* g2b   = (float*)alloc((size_t)N * 64 * 4);
  float* g3b   = (float*)alloc((size_t)N * 64 * 4);
  float* scvb  = (float*)alloc((size_t)N * 64 * 4);
  h16* wt1   = (h16*)alloc(4096 * 2);
  h16* wt2   = (h16*)alloc(8192 * 2);
  h16* wt3   = (h16*)alloc(8192 * 2);
  h16* wt4   = (h16*)alloc(8192 * 2);
  h16* wmixh = (h16*)alloc(24576 * 2);

  k_zero_out<<<1, 256, 0, stream>>>(out, N, G);
  k_init_node<<<(N + 255) / 256, 256, 0, stream>>>(attrs, E0, out, N, G);
  {
    size_t tot = (size_t)N * 576;
    k_init_feats<<<(int)((tot + 255) / 256), 256, 0, stream>>>(attrs, Wemb, feats, N);
  }
  k_edge<<<(E + 255) / 256, 256, 0, stream>>>(pos, eidx, sh, ef16, E);
  k_wconv<<<(53248 + 255) / 256, 256, 0, stream>>>(rW1, rW2, rW3, rW4, Wmix,
                                                   wt1, wt2, wt3, wt4, wmixh);

  for (int t = 0; t < 2; ++t) {
    k_zero_agg<<<2048, 256, 0, stream>>>(agg, (size_t)N * 576);
    int blocks_e = (E / 16) / 4;  // 4 waves x 16-edge tiles per block
    k_mlp_msg<<<blocks_e, 128, 0, stream>>>(ef16, sh, eidx, E, feats, agg,
                                            wt1 + (size_t)t * 2048, wt2 + (size_t)t * 4096,
                                            wt3 + (size_t)t * 4096, wt4 + (size_t)t * 4096);
    k_nodeW<<<(N * 64 + 255) / 256, 256, 0, stream>>>(attrs, w2, w3, Wsc, t, g2b, g3b, scvb, N);
    k_mix<<<(N / 16) / 4, 128, 0, stream>>>(agg, wmixh + (size_t)t * 12288);
    k_prod<<<(N * 64 + 255) / 256, 256, 0, stream>>>(agg, g2b, g3b, scvb, feats, N);
    k_readout<<<(N + 255) / 256, 256, 0, stream>>>(feats, Wr0, Wh, Wo, Wq, t, out, N, G);
  }
  k_coulomb<<<G, 256, 0, stream>>>(pos, out, N, G);
}
